// DIN_49838800503253
// MI455X (gfx1250) — matrix operand order repaired and checked by reading, not compiled, not run on hardware
//
#include <hip/hip_runtime.h>
#include <hip/hip_bf16.h>
#include <math.h>

// ---------------- problem constants ----------------
#define BB   8192
#define LL   40
#define NBF  2
#define VV   100000
#define EE   64
#define NSF  3
#define NDF  8
#define DD   128          // NB*E
#define ALLD 456          // D + D + (ND + NS*E)
#define NEG_INF_F (-4294967295.0f)

// WMMA vector types (probe-verified signatures)
typedef __attribute__((ext_vector_type(16))) _Float16 v16h;
typedef __attribute__((ext_vector_type(8)))  _Float16 v8h;
typedef __attribute__((ext_vector_type(8)))  float    v8f;

union AFrag { v16h v; v8h h[2]; };

// LDS strides (halves). Chosen for 16B alignment + bank spread.
#define AS   520     // info row stride: 512 + 8  (1040B, 16B aligned, banks spread by 4 dw)
#define H1S  104     // h1 row stride: 96 + 8     (208B, 16B aligned)
#define H2S  44

// workspace layout (bytes)
#define WS_W1PACK 0u          // 5*16*32*16 halves = 81920 B
#define WS_W2PACK 81920u      // 3*3*32*16 halves  = 9216 B
#define WS_W1P    91392u      // 456*80 f32        = 145920 B (BN-folded FFN W1)
#define WS_B1P    237312u     // 80 f32            = 320 B

// Fast sigmoid: native v_exp_f32 + native v_rcp_f32 (denominator >= 1,
// well conditioned; avoids the IEEE div_scale/div_fmas expansion).
__device__ __forceinline__ float sigmoidf_(float x) {
    return __builtin_amdgcn_rcpf(1.0f + __expf(-x));
}

// ------------------------------------------------------------------
// Prep kernel: pack attention weights into WMMA B-fragment layout,
// fold BatchNorm (inference affine) into FFN layer-1 weights/bias.
//
// B-fragment layout (16-bit B, 32x16, wave32): lane L holds column
// n = L%16; halves h=0..15 hold K = (L<16 ? 0 : 8) + (h&7) + (h&8 ? 16 : 0) of the k-tile.
// We store each fragment's per-lane 16 halves contiguously (32B/lane)
// so the GEMM loop does fully-coalesced 32B fragment loads.
// ------------------------------------------------------------------
__global__ void din_prep(const float* __restrict__ attW1,
                         const float* __restrict__ attW2,
                         const float* __restrict__ bn_g,
                         const float* __restrict__ bn_b,
                         const float* __restrict__ bn_m,
                         const float* __restrict__ bn_v,
                         const float* __restrict__ ffnW1,
                         const float* __restrict__ ffnb1,
                         _Float16* __restrict__ w1pack,
                         _Float16* __restrict__ w2pack,
                         float* __restrict__ w1p,
                         float* __restrict__ b1p)
{
    const int tid = threadIdx.x;
    const int nth = blockDim.x;

    // att_W1 [512,80] -> 5 n-tiles x 16 k-steps of packed fragments
    for (int e = tid; e < 5 * 16 * 32 * 16; e += nth) {
        int h    = e & 15;
        int lane = (e >> 4) & 31;
        int ks   = (e >> 9) & 15;
        int nt   = e >> 13;
        int n = nt * 16 + (lane & 15);
        int k = ks * 32 + ((lane & 16) ? 8 : 0) + (h & 7) + ((h & 8) ? 16 : 0);
        w1pack[e] = (_Float16)attW1[k * 80 + n];
    }
    // att_W2 [80,40] -> 3 n-tiles x 3 k-steps (zero-padded to 48x96)
    for (int e = tid; e < 3 * 3 * 32 * 16; e += nth) {
        int h    = e & 15;
        int lane = (e >> 4) & 31;
        int r    = e >> 9;           // 0..8
        int ks   = r % 3;
        int nt   = r / 3;
        int n = nt * 16 + (lane & 15);
        int k = ks * 32 + ((lane & 16) ? 8 : 0) + (h & 7) + ((h & 8) ? 16 : 0);
        w2pack[e] = (n < 40 && k < 80) ? (_Float16)attW2[k * 40 + n] : (_Float16)0.f;
    }
    // BN fold: y = g*(x-m)*rsqrt(v+eps)+beta ; y@W1+b1 == x@(s*W1) + b1'
    for (int e = tid; e < ALLD * 80; e += nth) {
        int i = e / 80;
        float s = bn_g[i] * rsqrtf(bn_v[i] + 1e-3f);
        w1p[e] = ffnW1[e] * s;
    }
    for (int t = tid; t < 80; t += nth) {
        float acc = ffnb1[t];
        for (int i = 0; i < ALLD; ++i) {
            float s = bn_g[i] * rsqrtf(bn_v[i] + 1e-3f);
            acc += (bn_b[i] - bn_m[i] * s) * ffnW1[i * 80 + t];
        }
        b1p[t] = acc;
    }
}

// ------------------------------------------------------------------
// Main fused DIN kernel: one block (4 wave32) per batch element.
// ------------------------------------------------------------------
__global__ __launch_bounds__(128)
void din_main(const float* __restrict__ dense_in,
              const int*   __restrict__ sparse_in,
              const int*   __restrict__ seq_in,
              const int*   __restrict__ item_in,
              const float* __restrict__ emb_sparse,
              const float* __restrict__ emb_seq,
              const _Float16* __restrict__ w1pack,
              const float* __restrict__ attb1,
              const _Float16* __restrict__ w2pack,
              const float* __restrict__ attb2,
              const float* __restrict__ attWf,
              const float* __restrict__ attbf,
              const float* __restrict__ w1p,   // BN-folded FFN W1 [456,80]
              const float* __restrict__ b1p,   // BN-folded FFN b1 [80]
              const float* __restrict__ ffna1,
              const float* __restrict__ ffnW2,
              const float* __restrict__ ffnb2,
              const float* __restrict__ ffna2,
              const float* __restrict__ outW,
              const float* __restrict__ outb,
              float* __restrict__ out)
{
    __shared__ __align__(16) _Float16 sInfo[48 * AS];   // attention input, f16
    __shared__ float   sSeq[LL * DD];                   // seq embeds, f32
    __shared__ __align__(16) _Float16 sH1[48 * H1S];    // sigmoid(info@W1+b1)
    __shared__ _Float16 sH2[48 * H2S];                  // sigmoid(h1@W2+b2)
    __shared__ float   sItem[DD];
    __shared__ float   sAll[ALLD];                      // [user|item|dense|sparse]
    __shared__ float   sLog[48];
    __shared__ float   sAtt[48];
    __shared__ int     sMask[48];
    __shared__ float   sHb[80];
    __shared__ float   sH2f[40];

    const int b    = blockIdx.x;
    const int tid  = threadIdx.x;
    const int wave = tid >> 5;
    const int lane = tid & 31;

    // ---- Phase 1: gathers (item, seq, dense, sparse) ----
    {
        const int i = tid >> 6, d = tid & 63;
        // item embedding (also goes directly into info_all slot)
        {
            int idx = item_in[(size_t)b * NBF + i];
            float v = emb_seq[((size_t)i * VV + idx) * EE + d];
            sItem[i * 64 + d] = v;
            sAll[DD + i * 64 + d] = v;
        }
        // behavior sequence embeddings
        for (int l = 0; l < LL; ++l) {
            int idx = seq_in[((size_t)b * LL + l) * NBF + i];
            sSeq[l * DD + i * 64 + d] = emb_seq[((size_t)i * VV + idx) * EE + d];
        }
        // masks
        if (tid < 48)
            sMask[tid] = (tid < LL) ? (seq_in[((size_t)b * LL + tid) * NBF] != 0) : 0;
        // dense features
        if (tid < NDF) sAll[2 * DD + tid] = dense_in[(size_t)b * NDF + tid];
        // other sparse embeddings
        for (int t = tid; t < NSF * EE; t += 128) {
            int si = t >> 6, dd = t & 63;
            int idx = sparse_in[(size_t)b * NSF + si];
            sAll[2 * DD + NDF + t] = emb_sparse[((size_t)si * VV + idx) * EE + dd];
        }
    }
    __syncthreads();

    // ---- Phase 2: build f16 attention input rows [q, s, q-s, q*s] ----
    for (int base = 0; base < LL; base += 2) {
        int l = base + (tid >> 6);
        int p = (tid & 63) * 2;
        float q0 = sItem[p], q1 = sItem[p + 1];
        float s0 = sSeq[l * DD + p], s1 = sSeq[l * DD + p + 1];
        _Float16* row = &sInfo[l * AS];
        row[p]           = (_Float16)q0;        row[p + 1]         = (_Float16)q1;
        row[DD + p]      = (_Float16)s0;        row[DD + p + 1]    = (_Float16)s1;
        row[2*DD + p]    = (_Float16)(q0 - s0); row[2*DD + p + 1]  = (_Float16)(q1 - s1);
        row[3*DD + p]    = (_Float16)(q0 * s0); row[3*DD + p + 1]  = (_Float16)(q1 * s1);
    }
    // zero padding rows 40..47
    for (int e = tid; e < 8 * AS; e += 128) sInfo[40 * AS + e] = (_Float16)0.f;
    __syncthreads();

    // ---- Phase 3: GEMM1  [48,512] x [512,80] -> sigmoid -> sH1 ----
    // 15 tiles (3 m-tiles x 5 n-tiles) over 4 waves; K loop = 16 WMMAs/tile.
    const int klo = (lane & 16) ? 8 : 0;   // A-frag lane k sub-offset
    for (int tile = wave; tile < 15; tile += 4) {
        const int mt = tile / 5, nt = tile % 5;
        const int mb = mt * 16, nb = nt * 16;
        v8f c = {0.f, 0.f, 0.f, 0.f, 0.f, 0.f, 0.f, 0.f};
        const _Float16* arow = &sInfo[(mb + (lane & 15)) * AS];
        const _Float16* bptr = w1pack + ((size_t)(nt * 16) * 32 + lane) * 16;
        #pragma unroll 4
        for (int ks = 0; ks < 16; ++ks) {
            const int kb = ks * 32;
            AFrag a;
            a.h[0] = *(const v8h*)(arow + kb + klo);        // K = kb+klo+0..7
            a.h[1] = *(const v8h*)(arow + kb + klo + 16);   // K = kb+16+klo+0..7
            v16h bf = *(const v16h*)(bptr);
            bptr += 32 * 16;
            c = __builtin_amdgcn_wmma_f32_16x16x32_f16(
                    false, a.v, false, bf, (short)0, c, false, false);
        }
        // epilogue: bias + sigmoid -> LDS f16
        const int nn = nb + (lane & 15);
        const float bias = attb1[nn];
        const int mo = mb + ((lane & 16) ? 8 : 0);
        #pragma unroll
        for (int r = 0; r < 8; ++r) {
            float x = c[r] + bias;
            sH1[(mo + r) * H1S + nn] = (_Float16)sigmoidf_(x);
        }
    }
    // zero K-pad columns 80..95 of sH1 (GEMM2 K is padded to 96)
    for (int e = tid; e < 48 * 16; e += 128) {
        int m = e >> 4, cc = 80 + (e & 15);
        sH1[m * H1S + cc] = (_Float16)0.f;
    }
    __syncthreads();

    // ---- Phase 4: GEMM2  [48,80(+pad)] x [80,40(+pad)] -> sigmoid -> sH2 ----
    for (int tile = wave; tile < 9; tile += 4) {
        const int mt = tile / 3, nt = tile % 3;
        const int mb = mt * 16, nb = nt * 16;
        v8f c = {0.f, 0.f, 0.f, 0.f, 0.f, 0.f, 0.f, 0.f};
        const _Float16* arow = &sH1[(mb + (lane & 15)) * H1S];
        const _Float16* bptr = w2pack + ((size_t)(nt * 3) * 32 + lane) * 16;
        #pragma unroll
        for (int ks = 0; ks < 3; ++ks) {
            const int kb = ks * 32;
            AFrag a;
            a.h[0] = *(const v8h*)(arow + kb + klo);
            a.h[1] = *(const v8h*)(arow + kb + klo + 16);
            v16h bf = *(const v16h*)(bptr + (size_t)ks * 32 * 16);
            c = __builtin_amdgcn_wmma_f32_16x16x32_f16(
                    false, a.v, false, bf, (short)0, c, false, false);
        }
        const int nn = nb + (lane & 15);
        if (nn < 40) {
            const float bias = attb2[nn];
            const int mo = mb + ((lane & 16) ? 8 : 0);
            #pragma unroll
            for (int r = 0; r < 8; ++r) {
                float x = c[r] + bias;
                sH2[(mo + r) * H2S + nn] = (_Float16)sigmoidf_(x);
            }
        }
    }
    __syncthreads();

    // ---- Phase 5: logits (40->1) + mask ----
    if (tid < 48) {
        float acc = attbf[0];
        #pragma unroll 8
        for (int j = 0; j < 40; ++j)
            acc += (float)sH2[tid * H2S + j] * attWf[j];
        sLog[tid] = (tid < LL && sMask[tid]) ? acc : NEG_INF_F;
    }
    __syncthreads();

    // ---- Phase 6: softmax over L=40 (tiny; serial on t0) ----
    if (tid == 0) {
        float mx = -3.4e38f;
        for (int l = 0; l < LL; ++l) mx = fmaxf(mx, sLog[l]);
        float sum = 0.f;
        for (int l = 0; l < LL; ++l) {
            float e = __expf(sLog[l] - mx);
            sAtt[l] = e;
            sum += e;
        }
        float inv = __builtin_amdgcn_rcpf(sum);
        for (int l = 0; l < LL; ++l) sAtt[l] *= inv;
    }
    __syncthreads();

    // ---- Phase 7: user_info = att @ seq_embed ----
    {
        float acc = 0.f;
        #pragma unroll 8
        for (int l = 0; l < LL; ++l) acc += sAtt[l] * sSeq[l * DD + tid];
        sAll[tid] = acc;
    }
    __syncthreads();

    // ---- Phase 8: FFN (BN pre-folded). Coalesced weight reads. ----
    if (tid < 80) {
        float acc = b1p[tid];
        for (int i = 0; i < ALLD; ++i) acc += sAll[i] * w1p[i * 80 + tid];
        sHb[tid] = (acc > 0.f) ? acc : ffna1[tid] * acc;   // PReLU
    }
    __syncthreads();
    if (tid < 40) {
        float acc = ffnb2[tid];
        #pragma unroll 8
        for (int j = 0; j < 80; ++j) acc += sHb[j] * ffnW2[j * 40 + tid];
        sH2f[tid] = (acc > 0.f) ? acc : ffna2[tid] * acc;  // PReLU
    }
    __syncthreads();
    if (tid == 0) {
        float z = outb[0];
        #pragma unroll 8
        for (int j = 0; j < 40; ++j) z += sH2f[j] * outW[j];
        out[b] = sigmoidf_(z);
    }
}

// ------------------------------------------------------------------
extern "C" void kernel_launch(void* const* d_in, const int* in_sizes, int n_in,
                              void* d_out, int out_size, void* d_ws, size_t ws_size,
                              hipStream_t stream) {
    const float* dense_in   = (const float*)d_in[0];
    const int*   sparse_in  = (const int*)  d_in[1];
    const int*   seq_in     = (const int*)  d_in[2];
    const int*   item_in    = (const int*)  d_in[3];
    const float* emb_sparse = (const float*)d_in[4];
    const float* emb_seq    = (const float*)d_in[5];
    const float* attW1      = (const float*)d_in[6];
    const float* attb1      = (const float*)d_in[7];
    const float* attW2      = (const float*)d_in[8];
    const float* attb2      = (const float*)d_in[9];
    const float* attWf      = (const float*)d_in[10];
    const float* attbf      = (const float*)d_in[11];
    const float* bn_g       = (const float*)d_in[12];
    const float* bn_b       = (const float*)d_in[13];
    const float* bn_m       = (const float*)d_in[14];
    const float* bn_v       = (const float*)d_in[15];
    const float* ffnW1      = (const float*)d_in[16];
    const float* ffnb1      = (const float*)d_in[17];
    const float* ffna1      = (const float*)d_in[18];
    const float* ffnW2      = (const float*)d_in[19];
    const float* ffnb2      = (const float*)d_in[20];
    const float* ffna2      = (const float*)d_in[21];
    const float* outW       = (const float*)d_in[22];
    const float* outb       = (const float*)d_in[23];
    float* out = (float*)d_out;

    char* ws = (char*)d_ws;
    _Float16* w1pack = (_Float16*)(ws + WS_W1PACK);
    _Float16* w2pack = (_Float16*)(ws + WS_W2PACK);
    float*    w1p    = (float*)   (ws + WS_W1P);
    float*    b1p    = (float*)   (ws + WS_B1P);

    din_prep<<<1, 256, 0, stream>>>(attW1, attW2, bn_g, bn_b, bn_m, bn_v,
                                    ffnW1, ffnb1, w1pack, w2pack, w1p, b1p);

    din_main<<<BB, 128, 0, stream>>>(dense_in, sparse_in, seq_in, item_in,
                                     emb_sparse, emb_seq,
                                     w1pack, attb1, w2pack, attb2, attWf, attbf,
                                     w1p, b1p, ffna1, ffnW2, ffnb2, ffna2,
                                     outW, outb, out);
}
